// FashionRGCN_65841848648447
// MI455X (gfx1250) — compile-verified, matrix operand order checked by reading, not run on hardware
//
#include <hip/hip_runtime.h>
#include <stdint.h>

#define NN 50000
#define NE 1600000
#define NR 8
#define NB 30
#define IN_CH 512
#define HID_CH 256
#define OUT_CH 128

typedef __attribute__((ext_vector_type(16))) __bf16 v16bf;
typedef __attribute__((ext_vector_type(8)))  float  v8f;

__device__ __forceinline__ unsigned short f2bf(float f) {
  unsigned int u = __float_as_uint(f);
  u += 0x7FFFu + ((u >> 16) & 1u);          // round-to-nearest-even
  return (unsigned short)(u >> 16);
}

// ---------------- f32 -> bf16 bulk convert (vectorized) ----------------
__global__ void cvt_f32_bf16(const float* __restrict__ in,
                             unsigned short* __restrict__ out, int n4) {
  int t = blockIdx.x * blockDim.x + threadIdx.x;
  if (t >= n4) return;
  float4 v = ((const float4*)in)[t];
  ushort4 o;
  o.x = f2bf(v.x); o.y = f2bf(v.y); o.z = f2bf(v.z); o.w = f2bf(v.w);
  ((ushort4*)out)[t] = o;
}

// ---- W[r] = sum_b comp[r,b]*bases[b], stored transposed bf16: Wt[(r*out+o)*K + i] ----
__global__ void build_w_bf16(const float* __restrict__ bases,
                             const float* __restrict__ comp,
                             unsigned short* __restrict__ Wt,
                             int inCh, int outCh) {
  int t = blockIdx.x * blockDim.x + threadIdx.x;
  if (t >= inCh * outCh) return;
  int i = t / outCh, o = t % outCh;
  float acc[NR];
#pragma unroll
  for (int r = 0; r < NR; ++r) acc[r] = 0.f;
  for (int b = 0; b < NB; ++b) {
    float bb = bases[((size_t)b * inCh + i) * outCh + o];
#pragma unroll
    for (int r = 0; r < NR; ++r) acc[r] += comp[r * NB + b] * bb;
  }
#pragma unroll
  for (int r = 0; r < NR; ++r)
    Wt[((size_t)(r * outCh + o)) * inCh + i] = f2bf(acc[r]);
}

// ---- rootT appended as extra N-columns: Wt[o*K + i] = root[i*out + o] ----
__global__ void transpose_bf16(const float* __restrict__ W,
                               unsigned short* __restrict__ Wt,
                               int inCh, int outCh) {
  int t = blockIdx.x * blockDim.x + threadIdx.x;
  if (t >= inCh * outCh) return;
  int i = t / outCh, o = t % outCh;
  Wt[(size_t)o * inCh + i] = f2bf(W[(size_t)i * outCh + o]);
}

// ---------------- bf16 WMMA GEMM: C[M,N] = A[M,K] * Bt[N,K]^T ----------------
// One wave computes a 16x64 strip of C (4 accumulators, A fragment reused 4x).
template <int K>
__global__ void gemm_bf16_wmma(const unsigned short* __restrict__ A,
                               const unsigned short* __restrict__ Bt,
                               float* __restrict__ C,
                               int nGroups, int totalJobs, int N) {
  int wave = blockIdx.x * (blockDim.x >> 5) + (threadIdx.x >> 5);
  if (wave >= totalJobs) return;             // wave-uniform: EXEC stays all-1s
  int mTile = wave / nGroups;
  int nGrp  = wave % nGroups;
  int lane  = threadIdx.x & 31;
  int half  = lane >> 4;                     // K-half select: half h holds k = 16h..16h+15 (contiguous)
  int l15   = lane & 15;

  const unsigned short* Ab = A  + (size_t)(mTile * 16 + l15) * K + half * 16;
  const unsigned short* Bb = Bt + (size_t)(nGrp  * 64 + l15) * K + half * 16;

  v8f c0 = {}, c1 = {}, c2 = {}, c3 = {};
  for (int k = 0; k < K; k += 32) {
    v16bf a  = *(const v16bf*)(Ab + k);
    v16bf b0 = *(const v16bf*)(Bb + k);
    v16bf b1 = *(const v16bf*)(Bb + 16 * K + k);
    v16bf b2 = *(const v16bf*)(Bb + 32 * K + k);
    v16bf b3 = *(const v16bf*)(Bb + 48 * K + k);
    __builtin_prefetch(Ab + k + 32, 0, 1);   // gfx1250 global_prefetch_b8
    c0 = __builtin_amdgcn_wmma_f32_16x16x32_bf16(false, a, false, b0, (short)0, c0, false, false);
    c1 = __builtin_amdgcn_wmma_f32_16x16x32_bf16(false, a, false, b1, (short)0, c1, false, false);
    c2 = __builtin_amdgcn_wmma_f32_16x16x32_bf16(false, a, false, b2, (short)0, c2, false, false);
    c3 = __builtin_amdgcn_wmma_f32_16x16x32_bf16(false, a, false, b3, (short)0, c3, false, false);
  }
  // C/D layout: lanes 0-15 -> rows m0..m0+7, lanes 16-31 -> rows m0+8..m0+15
  int crow = mTile * 16 + half * 8;
  int ccol = nGrp * 64 + l15;
#pragma unroll
  for (int i = 0; i < 8; ++i) {
    float* cp = C + (size_t)(crow + i) * N + ccol;
    cp[0] = c0[i]; cp[16] = c1[i]; cp[32] = c2[i]; cp[48] = c3[i];
  }
}

// ---------------- per-(tgt,rel) edge counts ----------------
__global__ void edge_count(const int* __restrict__ tgt, const int* __restrict__ et,
                           float* __restrict__ cnt, int E) {
  int e = blockIdx.x * blockDim.x + threadIdx.x;
  if (e >= E) return;
  atomicAdd(&cnt[tgt[e] * NR + et[e]], 1.0f);
}

// ---------------- gather H[src, etype] and scatter-add into sums[(tgt,rel)] ----------------
__global__ void edge_scatter(const float* __restrict__ H,
                             const int* __restrict__ src, const int* __restrict__ tgt,
                             const int* __restrict__ et,
                             float* __restrict__ sums,
                             int E, int C4, int C, int strideH) {
  int t = blockIdx.x * blockDim.x + threadIdx.x;
  int e = t / C4;
  if (e >= E) return;
  int c = (t % C4) * 4;
  int s = src[e], g = tgt[e], r = et[e];
  float4 v = *(const float4*)(H + (size_t)s * strideH + r * C + c);
  float* dp = sums + ((size_t)g * NR + r) * C + c;
  atomicAdd(dp + 0, v.x); atomicAdd(dp + 1, v.y);
  atomicAdd(dp + 2, v.z); atomicAdd(dp + 3, v.w);
}

// ---------------- layer-1 combine: mean over rel + root + bias, relu, -> bf16 ----------------
__global__ void combine_relu_bf16(const float* __restrict__ sums, const float* __restrict__ cnt,
                                  const float* __restrict__ H, const float* __restrict__ bias,
                                  unsigned short* __restrict__ out,
                                  int C, int strideH, int rootOff) {
  int t = blockIdx.x * blockDim.x + threadIdx.x;
  if (t >= NN * C) return;
  int n = t / C, c = t % C;
  float v = bias[c] + H[(size_t)n * strideH + rootOff + c];
#pragma unroll
  for (int r = 0; r < NR; ++r) {
    float cv = cnt[n * NR + r];
    v += sums[((size_t)n * NR + r) * C + c] / fmaxf(cv, 1.0f);
  }
  out[t] = f2bf(fmaxf(v, 0.0f));
}

// ---------------- layer-2 combine + L2 normalize (one node per 128-thread block) ----------------
__global__ void combine_norm(const float* __restrict__ sums, const float* __restrict__ cnt,
                             const float* __restrict__ H, const float* __restrict__ bias,
                             float* __restrict__ out) {
  int n = blockIdx.x, c = threadIdx.x;   // c in [0,128)
  float v = bias[c] + H[(size_t)n * (NR * OUT_CH + OUT_CH) + NR * OUT_CH + c];
#pragma unroll
  for (int r = 0; r < NR; ++r) {
    float cv = cnt[n * NR + r];
    v += sums[((size_t)n * NR + r) * OUT_CH + c] / fmaxf(cv, 1.0f);
  }
  __shared__ float red[OUT_CH];
  red[c] = v * v;
  __syncthreads();
  for (int s = OUT_CH / 2; s > 0; s >>= 1) {
    if (c < s) red[c] += red[c + s];
    __syncthreads();
  }
  float inv = 1.0f / fmaxf(sqrtf(red[0]), 1e-6f);
  out[(size_t)n * OUT_CH + c] = v * inv;
}

extern "C" void kernel_launch(void* const* d_in, const int* in_sizes, int n_in,
                              void* d_out, int out_size, void* d_ws, size_t ws_size,
                              hipStream_t stream) {
  const float* x      = (const float*)d_in[0];
  const int*   eidx   = (const int*)d_in[1];
  const int*   etype  = (const int*)d_in[2];
  const float* bases1 = (const float*)d_in[3];
  const float* comp1  = (const float*)d_in[4];
  const float* root1  = (const float*)d_in[5];
  const float* bias1  = (const float*)d_in[6];
  const float* bases2 = (const float*)d_in[7];
  const float* comp2  = (const float*)d_in[8];
  const float* root2  = (const float*)d_in[9];
  const float* bias2  = (const float*)d_in[10];
  const int* src = eidx;
  const int* tgt = eidx + NE;
  float* out = (float*)d_out;

  const int N1 = NR * HID_CH + HID_CH;   // 2304 (W1 cols + root1 cols)
  const int N2 = NR * OUT_CH + OUT_CH;   // 1152

  char* ws = (char*)d_ws;
  size_t off = 0;
  auto take = [&](size_t bytes) -> char* {
    char* p = ws + off;
    off = (off + bytes + 255) & ~(size_t)255;
    return p;
  };
  unsigned short* xb  = (unsigned short*)take((size_t)NN * IN_CH * 2);
  unsigned short* hb  = (unsigned short*)take((size_t)NN * HID_CH * 2);
  unsigned short* Wt1 = (unsigned short*)take((size_t)N1 * IN_CH * 2);
  unsigned short* Wt2 = (unsigned short*)take((size_t)N2 * HID_CH * 2);
  float* cnt  = (float*)take((size_t)NN * NR * 4);
  float* H    = (float*)take((size_t)NN * N1 * 4);            // reused for layer 2 (N2 < N1)
  float* sums = (float*)take((size_t)NN * NR * HID_CH * 4);   // reused for layer 2

  const int B = 256;

  // x -> bf16
  { int n4 = NN * IN_CH / 4;
    cvt_f32_bf16<<<(n4 + B - 1) / B, B, 0, stream>>>(x, xb, n4); }

  // layer-1 weights (transposed bf16) + rootT block
  { int n = IN_CH * HID_CH;
    build_w_bf16<<<(n + B - 1) / B, B, 0, stream>>>(bases1, comp1, Wt1, IN_CH, HID_CH);
    transpose_bf16<<<(n + B - 1) / B, B, 0, stream>>>(root1, Wt1 + (size_t)NR * HID_CH * IN_CH, IN_CH, HID_CH); }

  // GEMM 1: H = x @ [W1 | root1]   (M=50000, K=512, N=2304)
  { int nGrp = N1 / 64, jobs = (NN / 16) * nGrp;
    gemm_bf16_wmma<IN_CH><<<(jobs + 7) / 8, 256, 0, stream>>>(xb, Wt1, H, nGrp, jobs, N1); }

  // zero accumulators (workspace is not re-poisoned between replays)
  hipMemsetAsync(cnt, 0, (size_t)NN * NR * 4, stream);
  hipMemsetAsync(sums, 0, (size_t)NN * NR * HID_CH * 4, stream);

  edge_count<<<(NE + B - 1) / B, B, 0, stream>>>(tgt, etype, cnt, NE);

  { int tot = NE * (HID_CH / 4);
    edge_scatter<<<(tot + B - 1) / B, B, 0, stream>>>(H, src, tgt, etype, sums,
                                                      NE, HID_CH / 4, HID_CH, N1); }

  { int tot = NN * HID_CH;
    combine_relu_bf16<<<(tot + B - 1) / B, B, 0, stream>>>(sums, cnt, H, bias1, hb,
                                                           HID_CH, N1, NR * HID_CH); }

  // layer-2 weights
  { int n = HID_CH * OUT_CH;
    build_w_bf16<<<(n + B - 1) / B, B, 0, stream>>>(bases2, comp2, Wt2, HID_CH, OUT_CH);
    transpose_bf16<<<(n + B - 1) / B, B, 0, stream>>>(root2, Wt2 + (size_t)NR * OUT_CH * HID_CH, HID_CH, OUT_CH); }

  // GEMM 2: H = h @ [W2 | root2]   (M=50000, K=256, N=1152)
  { int nGrp = N2 / 64, jobs = (NN / 16) * nGrp;
    gemm_bf16_wmma<HID_CH><<<(jobs + 7) / 8, 256, 0, stream>>>(hb, Wt2, H, nGrp, jobs, N2); }

  hipMemsetAsync(sums, 0, (size_t)NN * NR * OUT_CH * 4, stream);

  { int tot = NE * (OUT_CH / 4);
    edge_scatter<<<(tot + B - 1) / B, B, 0, stream>>>(H, src, tgt, etype, sums,
                                                      NE, OUT_CH / 4, OUT_CH, N2); }

  combine_norm<<<NN, OUT_CH, 0, stream>>>(sums, cnt, H, bias2, out);
}
